// Transformer_19069654794586
// MI455X (gfx1250) — matrix operand order repaired and checked by reading, compiled, not run on hardware
//
#include <hip/hip_runtime.h>

// ---------------------------------------------------------------------------
// Transformer decoder block for MI455X (gfx1250, wave32, WMMA).
// B=32, T=1024, DIM=512, H=8, HD=64, HID=2048.
//
// Key algebraic simplification (exact): the self-attention einsums are
// degenerate -- softmax rows are contracted over their own axis, so the whole
// self-attention output is a single constant vector attnvec[512] broadcast
// over all (B,T) rows. Only the encoder-decoder attention and MLP are real
// GEMMs; those run on v_wmma_f32_16x16x32_bf16. bf16 A-tiles / Q-tiles are
// staged with GLOBAL_LOAD_ASYNC_TO_LDS_B128 (ASYNCcnt) -- the CDNA5 direct
// global->LDS path.
// ---------------------------------------------------------------------------

typedef __attribute__((ext_vector_type(16))) __bf16 v16bf;
typedef __attribute__((ext_vector_type(8)))  float  v8f;

union V16 { v16bf v; uint4 q[2]; };

__device__ __forceinline__ unsigned short f2bf(float f) {
    union { float f; unsigned u; } c; c.f = f;
    unsigned u = c.u;
    return (unsigned short)((u + 0x7FFFu + ((u >> 16) & 1u)) >> 16);  // RNE
}

// ---- CDNA5 async global->LDS copy (16 bytes), tracked by ASYNCcnt ---------
// GV mode: vdst = LDS byte address (low 32 bits of generic shared pointer,
// per the shared-aperture mapping); vaddr = 64-bit global addr.
__device__ __forceinline__ void async_copy16(const void* gsrc, void* ldsdst) {
    unsigned lds = (unsigned)(unsigned long long)ldsdst;
    asm volatile("global_load_async_to_lds_b128 %0, %1, off"
                 :: "v"(lds), "v"(gsrc)
                 : "memory");
}
__device__ __forceinline__ void async_wait0() {
    asm volatile("s_wait_asynccnt 0x0" ::: "memory");
}

// stage 16 contiguous A elements into LDS as bf16 (overloads pick input type)
__device__ __forceinline__ void stageA(unsigned short* dst, const unsigned short* src) {
    // bf16 input: direct async global->LDS, no VGPR round-trip
    async_copy16(src, dst);
    async_copy16(src + 8, dst + 8);
}
__device__ __forceinline__ void stageA(unsigned short* dst, const float* src) {
    const float4* s4 = reinterpret_cast<const float4*>(src);
#pragma unroll
    for (int i = 0; i < 4; ++i) {
        float4 f = s4[i];
        dst[i*4+0] = f2bf(f.x); dst[i*4+1] = f2bf(f.y);
        dst[i*4+2] = f2bf(f.z); dst[i*4+3] = f2bf(f.w);
    }
}

// ---------------------------------------------------------------------------
// Column-sum of x: xsum[c] = sum_{row} x[row][c]   (32768 rows x 512 cols)
// ---------------------------------------------------------------------------
__global__ __launch_bounds__(256) void colsum_kernel(const float* __restrict__ x,
                                                     float* __restrict__ xsum) {
    int t = threadIdx.x;
    size_t r0 = (size_t)blockIdx.x * 256;
    float a0 = 0.f, a1 = 0.f;
    for (int i = 0; i < 256; ++i) {
        const float* row = x + (r0 + i) * 512;
        a0 += row[t];
        a1 += row[t + 256];
    }
    atomicAdd(&xsum[t], a0);
    atomicAdd(&xsum[t + 256], a1);
}

// ---------------------------------------------------------------------------
// attnvec[c] = (tile8(SV) @ Wsa)[c] + bsa[c]
//   S[j]  = xsum . Wqkv[:,1024+j] + 32768*bqkv[1024+j]   (V-projection cols)
//   SV[f] = sum_h S[h*64+f]
// ---------------------------------------------------------------------------
__global__ __launch_bounds__(512) void attnvec_kernel(
        const float* __restrict__ xsum, const float* __restrict__ Wqkv,
        const float* __restrict__ bqkv, const float* __restrict__ Wsa,
        const float* __restrict__ bsa, float* __restrict__ attnvec) {
    __shared__ float S[512];
    __shared__ float SV[64];
    int j = threadIdx.x;
    float acc = 32768.0f * bqkv[1024 + j];
    for (int r = 0; r < 512; ++r) acc += xsum[r] * Wqkv[r * 1536 + 1024 + j];
    S[j] = acc;
    __syncthreads();
    if (j < 64) {
        float s = 0.f;
#pragma unroll
        for (int h = 0; h < 8; ++h) s += S[h * 64 + j];
        SV[j] = s;
    }
    __syncthreads();
    float a = bsa[j];
    for (int d = 0; d < 512; ++d) a += SV[d & 63] * Wsa[d * 512 + j];
    attnvec[j] = a;
}

// ---------------------------------------------------------------------------
// f32 -> bf16 conversion (weights)
// ---------------------------------------------------------------------------
__global__ __launch_bounds__(256) void f2bf_kernel(const float* __restrict__ s,
                                                   unsigned short* __restrict__ d, int n) {
    int i = blockIdx.x * 256 + threadIdx.x;
    if (i < n) d[i] = f2bf(s[i]);
}

// ---------------------------------------------------------------------------
// Row LayerNorm over 512 cols. Optional broadcast-add vector (attnvec),
// optional f32 and/or bf16 outputs.
// ---------------------------------------------------------------------------
__global__ __launch_bounds__(256) void ln_kernel(
        const float* __restrict__ src, const float* __restrict__ addvec,
        const float* __restrict__ g, const float* __restrict__ beta,
        float* __restrict__ outF, unsigned short* __restrict__ outB) {
    __shared__ float red[256];
    __shared__ float red2[256];
    __shared__ float s_mean, s_rstd;
    size_t row = blockIdx.x;
    int t = threadIdx.x;
    const float* s = src + row * 512;
    float v0 = s[t], v1 = s[t + 256];
    if (addvec) { v0 += addvec[t]; v1 += addvec[t + 256]; }
    red[t]  = v0 + v1;
    red2[t] = v0 * v0 + v1 * v1;
    __syncthreads();
    for (int step = 128; step > 0; step >>= 1) {
        if (t < step) { red[t] += red[t + step]; red2[t] += red2[t + step]; }
        __syncthreads();
    }
    if (t == 0) {
        float m = red[0] * (1.0f / 512.0f);
        float var = red2[0] * (1.0f / 512.0f) - m * m;
        s_mean = m;
        s_rstd = rsqrtf(var + 1e-6f);
    }
    __syncthreads();
    float m = s_mean, rs = s_rstd;
    float o0 = (v0 - m) * rs * g[t] + beta[t];
    float o1 = (v1 - m) * rs * g[t + 256] + beta[t + 256];
    if (outF) { outF[row * 512 + t] = o0; outF[row * 512 + t + 256] = o1; }
    if (outB) { outB[row * 512 + t] = f2bf(o0); outB[row * 512 + t + 256] = f2bf(o1); }
}

// ---------------------------------------------------------------------------
// WMMA GEMM:  C[M,N] = A[M,K] @ W[K,N]  (W pre-converted bf16, A bf16 or f32)
// Block tile 128x64, K-step 32, 256 threads (8 waves), wave tile 32x32
// (2x2 v_wmma_f32_16x16x32_bf16 accumulators).
// bf16 A-tiles staged via async global->LDS b128 copies (ASYNCcnt).
// Epilogues: 0 = +bias, bf16 scatter to qe[t,h,b,hd]
//            1 = +bias +res, f32 row-major            (z2 / z3)
//            2 = exact GELU(+bias), bf16 row-major    (hmid)
// ---------------------------------------------------------------------------
template <int EPI, typename AT>
__global__ __launch_bounds__(256) void gemm_wmma_kernel(
        const AT* __restrict__ A, const unsigned short* __restrict__ W,
        const float* __restrict__ bias, const float* __restrict__ res,
        float* __restrict__ outF, unsigned short* __restrict__ outB,
        int M, int N, int K) {
    __shared__ unsigned short As[128][32];     // bf16, 8 KB
    __shared__ unsigned short Bs[64][40];      // bf16 (n-major, padded), 5 KB

    const int tid  = threadIdx.x;
    const int lane = tid & 31;
    const int wave = tid >> 5;
    const int wm = (wave & 3) * 32;            // wave M offset in block tile
    const int wn = (wave >> 2) * 32;           // wave N offset in block tile
    const int blockRow = blockIdx.x * 128;
    const int blockCol = blockIdx.y * 64;

    v8f acc[2][2];
#pragma unroll
    for (int mi = 0; mi < 2; ++mi)
#pragma unroll
        for (int ni = 0; ni < 2; ++ni)
#pragma unroll
            for (int r = 0; r < 8; ++r) acc[mi][ni][r] = 0.0f;

    const int mlane  = lane & 15;
    const int kbaseA = (lane < 16) ? 0 : 8;    // A half-wave K interleave (ISA 7.12.2)
    const int kbaseB = (lane < 16) ? 0 : 8;    // B half-wave K interleave (matches A)

    for (int k0 = 0; k0 < K; k0 += 32) {
        // stage A tile 128x32 (2 threads per row, 16 elems each)
        {
            int r = tid >> 1;
            int c = (tid & 1) * 16;
            const AT* src = A + (size_t)(blockRow + r) * K + k0 + c;
            stageA(&As[r][c], src);
            if (k0 + 32 < K)
                __builtin_prefetch(src + 32, 0, 0);   // global_prefetch_b8 next K tile
        }
        // stage W tile 32x64, transposed into Bs[n][k]
        {
            int kk = tid >> 3;
            int n8 = (tid & 7) * 8;
            const unsigned short* src = W + (size_t)(k0 + kk) * N + blockCol + n8;
            uint4 d = *reinterpret_cast<const uint4*>(src);
            unsigned short e[8];
            *reinterpret_cast<uint4*>(e) = d;
#pragma unroll
            for (int j = 0; j < 8; ++j) Bs[n8 + j][kk] = e[j];
        }
        async_wait0();          // s_wait_asynccnt 0 (no-op for f32-A path)
        __syncthreads();

        v16bf af[2], bfg[2];
#pragma unroll
        for (int mi = 0; mi < 2; ++mi) {
            V16 u;
            int row = wm + mi * 16 + mlane;
            u.q[0] = *reinterpret_cast<const uint4*>(&As[row][kbaseA]);
            u.q[1] = *reinterpret_cast<const uint4*>(&As[row][kbaseA + 16]);
            af[mi] = u.v;
        }
#pragma unroll
        for (int ni = 0; ni < 2; ++ni) {
            V16 u;
            int n = wn + ni * 16 + mlane;
            u.q[0] = *reinterpret_cast<const uint4*>(&Bs[n][kbaseB]);
            u.q[1] = *reinterpret_cast<const uint4*>(&Bs[n][kbaseB + 16]);
            bfg[ni] = u.v;
        }
#pragma unroll
        for (int mi = 0; mi < 2; ++mi)
#pragma unroll
            for (int ni = 0; ni < 2; ++ni)
                acc[mi][ni] = __builtin_amdgcn_wmma_f32_16x16x32_bf16(
                    false, af[mi], false, bfg[ni], (short)0, acc[mi][ni], false, false);
        __syncthreads();
    }

    // epilogue; C layout: VGPR r -> m = r + 8*(lane/16), n = lane%16 (per tile)
#pragma unroll
    for (int mi = 0; mi < 2; ++mi) {
#pragma unroll
        for (int ni = 0; ni < 2; ++ni) {
            int n = blockCol + wn + ni * 16 + mlane;
#pragma unroll
            for (int r = 0; r < 8; ++r) {
                int row = blockRow + wm + mi * 16 + r + ((lane >> 4) << 3);
                float v = acc[mi][ni][r];
                if (EPI == 0) {               // qe scatter: [t,h,b,hd]
                    v += bias[n];
                    int b = row >> 10, t = row & 1023;
                    int h = n >> 6, hd = n & 63;
                    outB[(((size_t)(t * 8 + h) * 32 + b) << 6) + hd] = f2bf(v);
                } else if (EPI == 1) {        // z = acc + bias + residual (f32)
                    outF[(size_t)row * N + n] = v + bias[n] + res[(size_t)row * N + n];
                } else {                      // exact GELU -> bf16
                    float u = v + bias[n];
                    float gl = 0.5f * u * (1.0f + erff(u * 0.70710678118654752f));
                    outB[(size_t)row * N + n] = f2bf(gl);
                }
            }
        }
    }
}

// ---------------------------------------------------------------------------
// Encoder-decoder attention: one wave per (t,h) batch of 8192.
// Qe[32,64] (bf16) @ K[32,64]^T -> softmax -> @ V[32,64], all via WMMA.
// Qe tile staged via async global->LDS. Writes wa2 bf16 in [b, t, h*64+e].
// ---------------------------------------------------------------------------
__global__ __launch_bounds__(32) void attn_kernel(
        const unsigned short* __restrict__ qe, const float* __restrict__ kin,
        const float* __restrict__ vin, unsigned short* __restrict__ wa2) {
    __shared__ unsigned short qt[32][64];
    __shared__ unsigned short kt[32][64];
    __shared__ unsigned short vt[64][32];   // transposed V: vt[e][s]
    __shared__ float sc[32][32];
    __shared__ unsigned short pt[32][32];

    const int lane = threadIdx.x;
    const size_t base = (size_t)blockIdx.x * 32 * 64;   // (t,h) tile start
    const int t = blockIdx.x >> 3, h = blockIdx.x & 7;

    {   // Qe row (bf16 already): async global->LDS, 8 x b128 per lane
        const unsigned short* s = qe + base + lane * 64;
#pragma unroll
        for (int i = 0; i < 8; ++i)
            async_copy16(s + i * 8, &qt[lane][i * 8]);
    }
    {   // K row: f32 -> bf16
        const float4* s = reinterpret_cast<const float4*>(kin + base + lane * 64);
#pragma unroll
        for (int i = 0; i < 16; ++i) {
            float4 f = s[i];
            kt[lane][i*4+0] = f2bf(f.x); kt[lane][i*4+1] = f2bf(f.y);
            kt[lane][i*4+2] = f2bf(f.z); kt[lane][i*4+3] = f2bf(f.w);
        }
    }
    {   // V row -> transposed bf16
        const float4* s = reinterpret_cast<const float4*>(vin + base + lane * 64);
#pragma unroll
        for (int i = 0; i < 16; ++i) {
            float4 f = s[i];
            vt[i*4+0][lane] = f2bf(f.x); vt[i*4+1][lane] = f2bf(f.y);
            vt[i*4+2][lane] = f2bf(f.z); vt[i*4+3][lane] = f2bf(f.w);
        }
    }
    async_wait0();
    __syncthreads();

    const int mlane  = lane & 15;
    const int kbaseA = (lane < 16) ? 0 : 8;
    const int kbaseB = (lane < 16) ? 0 : 8;

    // dp = Qe @ K^T (K-dim = 64, two K-steps)
    v8f dp[2][2];
#pragma unroll
    for (int mi = 0; mi < 2; ++mi)
#pragma unroll
        for (int ni = 0; ni < 2; ++ni)
#pragma unroll
            for (int r = 0; r < 8; ++r) dp[mi][ni][r] = 0.0f;

#pragma unroll
    for (int ks = 0; ks < 64; ks += 32) {
        v16bf af[2], bfg[2];
#pragma unroll
        for (int mi = 0; mi < 2; ++mi) {
            V16 u;
            u.q[0] = *reinterpret_cast<const uint4*>(&qt[mi*16 + mlane][ks + kbaseA]);
            u.q[1] = *reinterpret_cast<const uint4*>(&qt[mi*16 + mlane][ks + kbaseA + 16]);
            af[mi] = u.v;
        }
#pragma unroll
        for (int ni = 0; ni < 2; ++ni) {
            V16 u;
            u.q[0] = *reinterpret_cast<const uint4*>(&kt[ni*16 + mlane][ks + kbaseB]);
            u.q[1] = *reinterpret_cast<const uint4*>(&kt[ni*16 + mlane][ks + kbaseB + 16]);
            bfg[ni] = u.v;
        }
#pragma unroll
        for (int mi = 0; mi < 2; ++mi)
#pragma unroll
            for (int ni = 0; ni < 2; ++ni)
                dp[mi][ni] = __builtin_amdgcn_wmma_f32_16x16x32_bf16(
                    false, af[mi], false, bfg[ni], (short)0, dp[mi][ni], false, false);
    }
#pragma unroll
    for (int mi = 0; mi < 2; ++mi)
#pragma unroll
        for (int ni = 0; ni < 2; ++ni)
#pragma unroll
            for (int r = 0; r < 8; ++r)
                sc[mi*16 + r + ((lane >> 4) << 3)][ni*16 + mlane] = dp[mi][ni][r] * 0.125f;
    __syncthreads();

    {   // softmax: lane owns row `lane`
        float ex[32];
        float mx = -3.4e38f;
#pragma unroll
        for (int j = 0; j < 32; ++j) { float v = sc[lane][j]; mx = fmaxf(mx, v); ex[j] = v; }
        float sum = 0.f;
#pragma unroll
        for (int j = 0; j < 32; ++j) { ex[j] = expf(ex[j] - mx); sum += ex[j]; }
        float inv = 1.0f / sum;
#pragma unroll
        for (int j = 0; j < 32; ++j) pt[lane][j] = f2bf(ex[j] * inv);
    }
    __syncthreads();

    // O = P @ V  (K-dim = 32, single step; N = 64 -> 4 tiles)
    v8f o[2][4];
#pragma unroll
    for (int mi = 0; mi < 2; ++mi)
#pragma unroll
        for (int ni = 0; ni < 4; ++ni)
#pragma unroll
            for (int r = 0; r < 8; ++r) o[mi][ni][r] = 0.0f;

    v16bf pf[2], vf[4];
#pragma unroll
    for (int mi = 0; mi < 2; ++mi) {
        V16 u;
        u.q[0] = *reinterpret_cast<const uint4*>(&pt[mi*16 + mlane][kbaseA]);
        u.q[1] = *reinterpret_cast<const uint4*>(&pt[mi*16 + mlane][kbaseA + 16]);
        pf[mi] = u.v;
    }
#pragma unroll
    for (int ni = 0; ni < 4; ++ni) {
        V16 u;
        u.q[0] = *reinterpret_cast<const uint4*>(&vt[ni*16 + mlane][kbaseB]);
        u.q[1] = *reinterpret_cast<const uint4*>(&vt[ni*16 + mlane][kbaseB + 16]);
        vf[ni] = u.v;
    }
#pragma unroll
    for (int mi = 0; mi < 2; ++mi)
#pragma unroll
        for (int ni = 0; ni < 4; ++ni)
            o[mi][ni] = __builtin_amdgcn_wmma_f32_16x16x32_bf16(
                false, pf[mi], false, vf[ni], (short)0, o[mi][ni], false, false);

#pragma unroll
    for (int mi = 0; mi < 2; ++mi)
#pragma unroll
        for (int ni = 0; ni < 4; ++ni)
#pragma unroll
            for (int r = 0; r < 8; ++r) {
                int b = mi*16 + r + ((lane >> 4) << 3);
                int e = ni*16 + mlane;
                wa2[((size_t)b * 1024 + t) * 512 + h * 64 + e] = f2bf(o[mi][ni][r]);
            }
}

// ---------------------------------------------------------------------------
// Host orchestration
// ---------------------------------------------------------------------------
extern "C" void kernel_launch(void* const* d_in, const int* in_sizes, int n_in,
                              void* d_out, int out_size, void* d_ws, size_t ws_size,
                              hipStream_t stream) {
    const float* x    = (const float*)d_in[0];
    const float* kin  = (const float*)d_in[1];
    const float* vin  = (const float*)d_in[2];
    const float* Wqkv = (const float*)d_in[3];
    const float* bqkv = (const float*)d_in[4];
    const float* Wsa  = (const float*)d_in[5];
    const float* bsa  = (const float*)d_in[6];
    const float* Wq   = (const float*)d_in[7];
    const float* bq   = (const float*)d_in[8];
    const float* Wed  = (const float*)d_in[9];
    const float* bed  = (const float*)d_in[10];
    const float* g1   = (const float*)d_in[11];
    const float* be1  = (const float*)d_in[12];
    const float* g2   = (const float*)d_in[13];
    const float* be2  = (const float*)d_in[14];
    const float* g3   = (const float*)d_in[15];
    const float* be3  = (const float*)d_in[16];
    const float* W1   = (const float*)d_in[17];
    const float* b1   = (const float*)d_in[18];
    const float* W2   = (const float*)d_in[19];
    const float* b2   = (const float*)d_in[20];
    float* out = (float*)d_out;

    const size_t ROWS = 32768;            // B*T
    char* w = (char*)d_ws;
    size_t off = 0;
    auto alloc = [&](size_t bytes) -> void* {
        void* p = w + off;
        off = (off + bytes + 255) & ~(size_t)255;
        return p;
    };
    float*          xsum    = (float*)alloc(512 * 4);
    float*          attnvec = (float*)alloc(512 * 4);
    unsigned short* Wq_bf   = (unsigned short*)alloc((size_t)512 * 512 * 2);
    unsigned short* Wed_bf  = (unsigned short*)alloc((size_t)512 * 512 * 2);
    unsigned short* W1_bf   = (unsigned short*)alloc((size_t)512 * 2048 * 2);
    unsigned short* W2_bf   = (unsigned short*)alloc((size_t)2048 * 512 * 2);
    unsigned short* an1_bf  = (unsigned short*)alloc(ROWS * 512 * 2);
    unsigned short* qe_bf   = (unsigned short*)alloc(ROWS * 512 * 2);
    unsigned short* wa2_bf  = (unsigned short*)alloc(ROWS * 512 * 2);
    float*          zbuf    = (float*)alloc(ROWS * 512 * 4);   // z2 then z3
    float*          an2     = (float*)alloc(ROWS * 512 * 4);
    unsigned short* hmid    = (unsigned short*)alloc((size_t)8192 * 2048 * 2);

    // 1) constant self-attention vector
    hipMemsetAsync(xsum, 0, 512 * sizeof(float), stream);
    colsum_kernel<<<128, 256, 0, stream>>>(x, xsum);
    attnvec_kernel<<<1, 512, 0, stream>>>(xsum, Wqkv, bqkv, Wsa, bsa, attnvec);

    // 2) weight conversions to bf16
    f2bf_kernel<<<(512 * 512 + 255) / 256, 256, 0, stream>>>(Wq, Wq_bf, 512 * 512);
    f2bf_kernel<<<(512 * 512 + 255) / 256, 256, 0, stream>>>(Wed, Wed_bf, 512 * 512);
    f2bf_kernel<<<(512 * 2048 + 255) / 256, 256, 0, stream>>>(W1, W1_bf, 512 * 2048);
    f2bf_kernel<<<(2048 * 512 + 255) / 256, 256, 0, stream>>>(W2, W2_bf, 2048 * 512);

    // 3) an1 = LN(x + attnvec) -> bf16
    ln_kernel<<<ROWS, 256, 0, stream>>>(x, attnvec, g1, be1, nullptr, an1_bf);

    // 4) qe = an1 @ Wq + bq, scattered to [t,h,b,hd]
    gemm_wmma_kernel<0, unsigned short><<<dim3(256, 8), 256, 0, stream>>>(
        an1_bf, Wq_bf, bq, nullptr, nullptr, qe_bf, 32768, 512, 512);

    // 5) batched encoder-decoder attention
    attn_kernel<<<8192, 32, 0, stream>>>(qe_bf, kin, vin, wa2_bf);

    // 6) z2 = wa2 @ Wed + bed + x ; an2 = LN(z2)
    gemm_wmma_kernel<1, unsigned short><<<dim3(256, 8), 256, 0, stream>>>(
        wa2_bf, Wed_bf, bed, x, zbuf, nullptr, 32768, 512, 512);
    ln_kernel<<<ROWS, 256, 0, stream>>>(zbuf, nullptr, g2, be2, an2, nullptr);

    // 7) MLP in 4 row-chunks of 8192 (bounds hmid workspace)
    for (int c = 0; c < 4; ++c) {
        size_t ro = (size_t)c * 8192;
        gemm_wmma_kernel<2, float><<<dim3(64, 32), 256, 0, stream>>>(
            an2 + ro * 512, W1_bf, b1, nullptr, nullptr, hmid, 8192, 2048, 512);
        gemm_wmma_kernel<1, unsigned short><<<dim3(64, 8), 256, 0, stream>>>(
            hmid, W2_bf, b2, an2 + ro * 512, zbuf + ro * 512, nullptr, 8192, 512, 2048);
    }

    // 8) out = LN(z3) with z3 = hmid@W2 + b2 + an2 already in zbuf
    ln_kernel<<<ROWS, 256, 0, stream>>>(zbuf, nullptr, g3, be3, out, nullptr);
}
